// AdaptiveSoftmax_48361331753331
// MI455X (gfx1250) — compile-verified, matrix operand order checked by reading, not run on hardware
//
#include <hip/hip_runtime.h>

#define N_ROWS   4096
#define HDIM     1024
#define CUT0     4000
#define CUT1     20000
#define VOCAB    50257
#define HEAD_OUT 4002
#define P0       1024
#define P1       256

#if __has_builtin(__builtin_amdgcn_global_load_async_to_lds_b128) && \
    __has_builtin(__builtin_amdgcn_s_wait_asynccnt)
#define HAVE_ASYNC 1
#else
#define HAVE_ASYNC 0
#endif

typedef __attribute__((ext_vector_type(16))) __bf16 v16bf;
typedef __attribute__((ext_vector_type(8)))  float  v8f;
typedef int v4i_t __attribute__((vector_size(16)));   // GCC-style, matches builtin param pointee

#define BM 64
#define BN 128
#define BK 32
#define LDT 40   // LDS row stride in bf16 elems: 80B = 5x16B (keeps B128 align, spreads banks)

struct chunk16 { unsigned int w[4]; };   // 16-byte tile chunk

#if HAVE_ASYNC
__device__ __forceinline__ void async_cp16(const __bf16* g, __bf16* l) {
  __builtin_amdgcn_global_load_async_to_lds_b128(
      (__attribute__((address_space(1))) v4i_t*)g,
      (__attribute__((address_space(3))) v4i_t*)l, 0, 0);
}
#endif

// C[m][n] = sum_k X[m][k]*W[n][k] (+bias[n]); X,W pre-converted bf16, K-contiguous.
// Output: fp32 (outf) or bf16 (outb). Head mode (cls): cols >= CUT0 -> cls buffer.
__global__ __launch_bounds__(256) void gemm_wmma_bf16(
    const __bf16* __restrict__ X, const __bf16* __restrict__ W,
    const float* __restrict__ bias, float* __restrict__ outf,
    __bf16* __restrict__ outb, float* __restrict__ cls,
    int K, int Ncols, int out_stride)
{
  __shared__ __bf16 As[2][BM][LDT];
  __shared__ __bf16 Bs[2][BN][LDT];

  const int bm   = blockIdx.y * BM;
  const int bn   = blockIdx.x * BN;
  const int t    = threadIdx.x;      // 256 = 8 wave32
  const int wave = t >> 5;
  const int lane = t & 31;
  const int wm   = (wave & 1) * 32;
  const int wn   = (wave >> 1) * 32;
  const int lrow = lane & 15;
  const int kb   = (lane >> 4) * 8;  // lanes 0-15: K 0..7/16..23; lanes 16-31: K 8..15/24..31

  v8f acc[2][2] = {};

  // tile copy assignments: A 64x32 -> 1 chunk/thread, B 128x32 -> 2 chunks/thread
  const int ar = t >> 2, ac = (t & 3) * 8;
  const int br = t >> 1, bc = (t & 1) * 16;
  const __bf16* ag = X + (size_t)(bm + ar) * K + ac;
  const int     wr = bn + br;
  const __bf16* bg = W + (size_t)wr * K + bc;
  const bool bvalid = (wr < Ncols);

  // zero-fill out-of-range B rows once (never async-written afterwards)
  if (!bvalid) {
    #pragma unroll
    for (int b = 0; b < 2; ++b)
      #pragma unroll
      for (int i = 0; i < 16; ++i) Bs[b][br][bc + i] = (__bf16)0.0f;
  }

  // prologue: stage k0 = 0 into buffer 0
#if HAVE_ASYNC
  async_cp16(ag, &As[0][ar][ac]);
  if (bvalid) { async_cp16(bg, &Bs[0][br][bc]); async_cp16(bg + 8, &Bs[0][br][bc + 8]); }
#else
  *(chunk16*)&As[0][ar][ac] = *(const chunk16*)ag;
  if (bvalid) {
    *(chunk16*)&Bs[0][br][bc]     = *(const chunk16*)bg;
    *(chunk16*)&Bs[0][br][bc + 8] = *(const chunk16*)(bg + 8);
  }
#endif

  int cur = 0;
  for (int k0 = 0; k0 < K; k0 += BK) {
#if HAVE_ASYNC
    __builtin_amdgcn_s_wait_asynccnt(0);   // this wave's LDS DMA done
#endif
    __syncthreads();                       // tile[cur] visible; all waves done reading tile[cur^1]

    if (k0 + BK < K) {                     // stage next tile into the other buffer
      const int nxt = cur ^ 1, kn = k0 + BK;
#if HAVE_ASYNC
      async_cp16(ag + kn, &As[nxt][ar][ac]);
      if (bvalid) { async_cp16(bg + kn, &Bs[nxt][br][bc]);
                    async_cp16(bg + kn + 8, &Bs[nxt][br][bc + 8]); }
#else
      *(chunk16*)&As[nxt][ar][ac] = *(const chunk16*)(ag + kn);
      if (bvalid) {
        *(chunk16*)&Bs[nxt][br][bc]     = *(const chunk16*)(bg + kn);
        *(chunk16*)&Bs[nxt][br][bc + 8] = *(const chunk16*)(bg + kn + 8);
      }
#endif
    }

    v16bf afrag[2], bfrag[2];
    #pragma unroll
    for (int tm = 0; tm < 2; ++tm) {
      const __bf16* ap = &As[cur][wm + tm * 16 + lrow][kb];
      #pragma unroll
      for (int i = 0; i < 8; ++i) { afrag[tm][i] = ap[i]; afrag[tm][8 + i] = ap[16 + i]; }
    }
    #pragma unroll
    for (int tn = 0; tn < 2; ++tn) {
      const __bf16* bp = &Bs[cur][wn + tn * 16 + lrow][kb];
      #pragma unroll
      for (int i = 0; i < 8; ++i) { bfrag[tn][i] = bp[i]; bfrag[tn][8 + i] = bp[16 + i]; }
    }
    #pragma unroll
    for (int tm = 0; tm < 2; ++tm)
      #pragma unroll
      for (int tn = 0; tn < 2; ++tn)
        acc[tm][tn] = __builtin_amdgcn_wmma_f32_16x16x32_bf16(
            false, afrag[tm], false, bfrag[tn], (short)0, acc[tm][tn], false, false);
    cur ^= 1;
  }

  // C/D layout: VGPR e, lanes 0-15 -> M=e, lanes 16-31 -> M=e+8; N = lane&15
  const int lhalf = lane >> 4;
  #pragma unroll
  for (int tm = 0; tm < 2; ++tm) {
    #pragma unroll
    for (int tn = 0; tn < 2; ++tn) {
      const int n = bn + wn + tn * 16 + lrow;
      if (n >= Ncols) continue;
      const float bv = bias ? bias[n] : 0.0f;
      #pragma unroll
      for (int e = 0; e < 8; ++e) {
        const int m = bm + wm + tm * 16 + lhalf * 8 + e;
        const float v = acc[tm][tn][e] + bv;
        if (cls && n >= CUT0)      cls[m * 2 + (n - CUT0)] = v;
        else if (outb)             outb[(size_t)m * out_stride + n] = (__bf16)v;
        else                       outf[(size_t)m * out_stride + n] = v;
      }
    }
  }
}

// elementwise fp32 -> bf16 (8 elems/thread)
__global__ __launch_bounds__(256) void f32_to_bf16(
    const float* __restrict__ in, __bf16* __restrict__ out, int n8)
{
  const int i = blockIdx.x * 256 + threadIdx.x;
  if (i >= n8) return;
  const size_t base = (size_t)i * 8;
  #pragma unroll
  for (int j = 0; j < 8; ++j) out[base + j] = (__bf16)in[base + j];
}

// log-softmax over 4002-wide head row (4000 cols in d_out, 2 cluster cols in ws)
__global__ __launch_bounds__(256) void head_softmax(
    float* __restrict__ out, const float* __restrict__ cls_raw,
    float* __restrict__ cls_lp)
{
  const int r = blockIdx.x;
  float* row = out + (size_t)r * VOCAB;
  __shared__ float red[256];
  const int t = threadIdx.x;

  float m = -3.0e38f;
  for (int j = t; j < CUT0; j += 256) m = fmaxf(m, row[j]);
  if (t < 2) m = fmaxf(m, cls_raw[r * 2 + t]);
  red[t] = m; __syncthreads();
  for (int s = 128; s > 0; s >>= 1) { if (t < s) red[t] = fmaxf(red[t], red[t + s]); __syncthreads(); }
  m = red[0]; __syncthreads();

  float sum = 0.0f;
  for (int j = t; j < CUT0; j += 256) sum += __expf(row[j] - m);
  if (t < 2) sum += __expf(cls_raw[r * 2 + t] - m);
  red[t] = sum; __syncthreads();
  for (int s = 128; s > 0; s >>= 1) { if (t < s) red[t] += red[t + s]; __syncthreads(); }

  const float lse = m + __logf(red[0]);
  for (int j = t; j < CUT0; j += 256) row[j] -= lse;
  if (t < 2) cls_lp[r * 2 + t] = cls_raw[r * 2 + t] - lse;
}

// in-place log-softmax over a tail span + broadcast-add of cluster log-prob
__global__ __launch_bounds__(512) void tail_softmax(
    float* __restrict__ out, const float* __restrict__ cls_lp,
    int col_off, int count, int cls_idx)
{
  const int r = blockIdx.x;
  float* row = out + (size_t)r * VOCAB + col_off;
  __shared__ float red[512];
  const int t = threadIdx.x;

  float m = -3.0e38f;
  for (int j = t; j < count; j += 512) m = fmaxf(m, row[j]);
  red[t] = m; __syncthreads();
  for (int s = 256; s > 0; s >>= 1) { if (t < s) red[t] = fmaxf(red[t], red[t + s]); __syncthreads(); }
  m = red[0]; __syncthreads();

  float sum = 0.0f;
  for (int j = t; j < count; j += 512) sum += __expf(row[j] - m);
  red[t] = sum; __syncthreads();
  for (int s = 256; s > 0; s >>= 1) { if (t < s) red[t] += red[t + s]; __syncthreads(); }

  const float adj = cls_lp[r * 2 + cls_idx] - (m + __logf(red[0]));
  for (int j = t; j < count; j += 512) row[j] += adj;
}

extern "C" void kernel_launch(void* const* d_in, const int* in_sizes, int n_in,
                              void* d_out, int out_size, void* d_ws, size_t ws_size,
                              hipStream_t stream) {
  const float* x      = (const float*)d_in[0];
  const float* head_w = (const float*)d_in[1];
  const float* head_b = (const float*)d_in[2];
  const float* t0_w1  = (const float*)d_in[3];
  const float* t0_w2  = (const float*)d_in[4];
  const float* t1_w1  = (const float*)d_in[5];
  const float* t1_w2  = (const float*)d_in[6];
  float* out = (float*)d_out;

  // bf16 workspace segments (all sizes multiples of 8 elems -> 16B aligned)
  __bf16* xb    = (__bf16*)d_ws;
  __bf16* hwb   = xb    + (size_t)N_ROWS * HDIM;
  __bf16* t0w1b = hwb   + (size_t)HEAD_OUT * HDIM;
  __bf16* t0w2b = t0w1b + (size_t)P0 * HDIM;
  __bf16* t1w1b = t0w2b + (size_t)(CUT1 - CUT0) * P0;
  __bf16* t1w2b = t1w1b + (size_t)P1 * HDIM;
  __bf16* h0b   = t1w2b + (size_t)(VOCAB - CUT1) * P1;
  __bf16* h1b   = h0b   + (size_t)N_ROWS * P0;
  float* cls_raw = (float*)(h1b + (size_t)N_ROWS * P1);
  float* cls_lp  = cls_raw + (size_t)N_ROWS * 2;

  // one-time fp32 -> bf16 conversion of activations + all weights (~220MB, ~10us)
  {
    struct { const float* in; __bf16* out; size_t n; } jobs[6] = {
      { x,      xb,    (size_t)N_ROWS * HDIM },
      { head_w, hwb,   (size_t)HEAD_OUT * HDIM },
      { t0_w1,  t0w1b, (size_t)P0 * HDIM },
      { t0_w2,  t0w2b, (size_t)(CUT1 - CUT0) * P0 },
      { t1_w1,  t1w1b, (size_t)P1 * HDIM },
      { t1_w2,  t1w2b, (size_t)(VOCAB - CUT1) * P1 },
    };
    for (int j = 0; j < 6; ++j) {
      const int n8 = (int)(jobs[j].n / 8);
      f32_to_bf16<<<(n8 + 255) / 256, 256, 0, stream>>>(jobs[j].in, jobs[j].out, n8);
    }
  }

  const dim3 blk(256);
  // h0 = x @ t0_w1^T  -> bf16 [4096 x 1024]
  gemm_wmma_bf16<<<dim3(P0 / BN, N_ROWS / BM), blk, 0, stream>>>(
      xb, t0w1b, nullptr, nullptr, h0b, nullptr, HDIM, P0, P0);
  // h1 = x @ t1_w1^T  -> bf16 [4096 x 256]
  gemm_wmma_bf16<<<dim3((P1 + BN - 1) / BN, N_ROWS / BM), blk, 0, stream>>>(
      xb, t1w1b, nullptr, nullptr, h1b, nullptr, HDIM, P1, P1);
  // head logits -> d_out[:, 0:4000] (+bias); cluster cols -> cls_raw
  gemm_wmma_bf16<<<dim3((HEAD_OUT + BN - 1) / BN, N_ROWS / BM), blk, 0, stream>>>(
      xb, hwb, head_b, out, nullptr, cls_raw, HDIM, HEAD_OUT, VOCAB);
  // tail0 logits -> d_out[:, 4000:20000]
  gemm_wmma_bf16<<<dim3((CUT1 - CUT0 + BN - 1) / BN, N_ROWS / BM), blk, 0, stream>>>(
      h0b, t0w2b, nullptr, out + CUT0, nullptr, nullptr, P0, CUT1 - CUT0, VOCAB);
  // tail1 logits -> d_out[:, 20000:50257]
  gemm_wmma_bf16<<<dim3((VOCAB - CUT1 + BN - 1) / BN, N_ROWS / BM), blk, 0, stream>>>(
      h1b, t1w2b, nullptr, out + CUT1, nullptr, nullptr, P1, VOCAB - CUT1, VOCAB);

  head_softmax<<<N_ROWS, 256, 0, stream>>>(out, cls_raw, cls_lp);
  tail_softmax<<<N_ROWS, 512, 0, stream>>>(out, cls_lp, CUT0, CUT1 - CUT0, 0);
  tail_softmax<<<N_ROWS, 512, 0, stream>>>(out, cls_lp, CUT1, VOCAB - CUT1, 1);
}
